// RSNN_LSTM_4398046511945
// MI455X (gfx1250) — matrix operand order repaired and checked by reading, not compiled, not run on hardware
//
#include <hip/hip_runtime.h>

typedef _Float16 v16h __attribute__((ext_vector_type(16)));
typedef _Float16 v8h  __attribute__((ext_vector_type(8)));
typedef float    v8f  __attribute__((ext_vector_type(8)));

namespace {
constexpr int T_ = 128, B_ = 256, NI_ = 1024, NH_ = 2048, NO_ = 256;
constexpr long M_   = (long)T_ * B_;     // 32768 batched rows
constexpr long TBNH = M_ * NH_;          // 67,108,864
constexpr long TBNO = M_ * NO_;          // 8,388,608
}

// ---------------------------------------------------------------- conversion
__global__ void cvt_f32_f16(const float* __restrict__ s, _Float16* __restrict__ d, long n) {
    long i = ((long)blockIdx.x * blockDim.x + threadIdx.x) * 8;
    if (i >= n) return;
    float4 a = *(const float4*)(s + i);
    float4 b = *(const float4*)(s + i + 4);
    v8h h = { (_Float16)a.x, (_Float16)a.y, (_Float16)a.z, (_Float16)a.w,
              (_Float16)b.x, (_Float16)b.y, (_Float16)b.z, (_Float16)b.w };
    *(v8h*)(d + i) = h;
}

// A-fragment loader: 16x32 A in contiguous k order, as B uses (lanes 0-15: K=0..15; lanes 16-31: K=16..31)
__device__ __forceinline__ v16h load_afrag(const _Float16* __restrict__ p) {
    union { v16h v; v8h h[2]; } u;
    u.h[0] = *(const v8h*)(p);
    u.h[1] = *(const v8h*)(p + 8);
    return u.v;
}

// ------------------------------------------- fused dual GEMM: cand & forget
// cand = sigmoid(X W1^T + b1), forget = relu(X Wr^T + br)   [M_ x NH_]
__global__ void __launch_bounds__(256)
gemm_dual_f16(const _Float16* __restrict__ Xh, const _Float16* __restrict__ W1h,
              const _Float16* __restrict__ Wrh, const float* __restrict__ b1,
              const float* __restrict__ br, float* __restrict__ cand,
              float* __restrict__ forg) {
    constexpr int K = NI_, N = NH_;
    constexpr int NT = N / 16;                       // 128 n-tiles
    int wid  = blockIdx.x * 8 + (threadIdx.x >> 5);  // global wave id
    int lane = threadIdx.x & 31;
    int ll = lane & 15, lh = lane >> 4;
    int nbase  = (wid % NT) * 16;
    long mbase = (long)(wid / NT) * 64;              // 4 M-tiles per wave

    const _Float16* bp1 = W1h + (long)(nbase + ll) * K + lh * 16;  // B = W^T col-major tile
    const _Float16* bp2 = Wrh + (long)(nbase + ll) * K + lh * 16;
    const _Float16* ap  = Xh + (mbase + ll) * K + lh * 16;

    const v8f zero = {0.f,0.f,0.f,0.f,0.f,0.f,0.f,0.f};
    v8f acc1[4] = {zero, zero, zero, zero};
    v8f acc2[4] = {zero, zero, zero, zero};

    for (int kk = 0; kk < K; kk += 32) {
        v16h bf1 = *(const v16h*)(bp1 + kk);
        v16h bf2 = *(const v16h*)(bp2 + kk);
#pragma unroll
        for (int i = 0; i < 4; ++i) {
            v16h af = load_afrag(ap + (long)i * 16 * K + kk);
            acc1[i] = __builtin_amdgcn_wmma_f32_16x16x32_f16(
                false, af, false, bf1, (short)0, acc1[i], false, false);
            acc2[i] = __builtin_amdgcn_wmma_f32_16x16x32_f16(
                false, af, false, bf2, (short)0, acc2[i], false, false);
        }
    }

    int col = nbase + ll;
    float bb1 = b1[col], bbr = br[col];
#pragma unroll
    for (int i = 0; i < 4; ++i) {
#pragma unroll
        for (int r = 0; r < 8; ++r) {
            long row = mbase + i * 16 + lh * 8 + r;   // C/D layout: lanes16-31 hold M+8
            float t1 = acc1[i][r] + bb1;
            float t2 = acc2[i][r] + bbr;
            cand[row * N + col] = 1.0f / (1.0f + __expf(-t1));
            forg[row * N + col] = t2 > 0.f ? t2 : 0.f;
        }
    }
}

// ------------------------------------------------ GEMM: inp2 = spk1 W2^T + b2
__global__ void __launch_bounds__(256)
gemm_out_f16(const _Float16* __restrict__ Ah, const _Float16* __restrict__ W2h,
             const float* __restrict__ b2, float* __restrict__ outp) {
    constexpr int K = NH_, N = NO_;
    constexpr int NT = N / 16;                       // 16 n-tiles
    int wid  = blockIdx.x * 8 + (threadIdx.x >> 5);
    int lane = threadIdx.x & 31;
    int ll = lane & 15, lh = lane >> 4;
    int nbase  = (wid % NT) * 16;
    long mbase = (long)(wid / NT) * 64;

    const _Float16* bp = W2h + (long)(nbase + ll) * K + lh * 16;
    const _Float16* ap = Ah + (mbase + ll) * K + lh * 16;

    const v8f zero = {0.f,0.f,0.f,0.f,0.f,0.f,0.f,0.f};
    v8f acc[4] = {zero, zero, zero, zero};

    for (int kk = 0; kk < K; kk += 32) {
        v16h bf = *(const v16h*)(bp + kk);
#pragma unroll
        for (int i = 0; i < 4; ++i) {
            v16h af = load_afrag(ap + (long)i * 16 * K + kk);
            acc[i] = __builtin_amdgcn_wmma_f32_16x16x32_f16(
                false, af, false, bf, (short)0, acc[i], false, false);
        }
    }

    int col = nbase + ll;
    float bb = b2[col];
#pragma unroll
    for (int i = 0; i < 4; ++i) {
#pragma unroll
        for (int r = 0; r < 8; ++r) {
            long row = mbase + i * 16 + lh * 8 + r;
            outp[row * N + col] = acc[i][r] + bb;
        }
    }
}

// ------------------------------------------- layer-1 recurrence (per (b,h))
__global__ void rec1_kernel(const float* __restrict__ cand, const float* __restrict__ forg,
                            float* __restrict__ spk1_rec, float* __restrict__ mem1_rec,
                            _Float16* __restrict__ spk1h) {
    long idx = (long)blockIdx.x * blockDim.x + threadIdx.x;   // [0, B_*NH_)
    const long stride = (long)B_ * NH_;
    float syn = 0.f, mem = 0.f;
    long off = idx;
    for (int t = 0; t < T_; ++t, off += stride) {
        float c = cand[off], f = forg[off];
        syn = f * syn + (1.f - f) * c;                // syn update BEFORE spike
        float spk = (mem - 1.0f) > 0.f ? 1.f : 0.f;   // spike from OLD mem
        mem = 0.5f * mem + syn - spk;                 // BETA=0.5, THR=1
        spk1_rec[off] = spk;
        mem1_rec[off] = mem;
        spk1h[off]    = (_Float16)spk;                // exact 0/1 for GEMM2
    }
}

// ------------------------------------------- layer-2 recurrence (per (b,o))
__global__ void rec2_kernel(const float* __restrict__ inp2,
                            float* __restrict__ spk2_rec, float* __restrict__ mem2_rec) {
    long idx = (long)blockIdx.x * blockDim.x + threadIdx.x;   // [0, B_*NO_)
    const long stride = (long)B_ * NO_;
    float syn = 0.f, mem = 0.f;
    long off = idx;
    for (int t = 0; t < T_; ++t, off += stride) {
        float i2 = inp2[off];
        float spk = (mem - 1.0f) > 0.f ? 1.f : 0.f;   // spike from OLD mem
        syn = 0.5f * syn + i2;
        mem = 0.5f * mem + syn - spk;
        spk2_rec[off] = spk;
        mem2_rec[off] = mem;
    }
}

extern "C" void kernel_launch(void* const* d_in, const int* in_sizes, int n_in,
                              void* d_out, int out_size, void* d_ws, size_t ws_size,
                              hipStream_t stream) {
    const float* x  = (const float*)d_in[0];
    const float* W1 = (const float*)d_in[1];
    const float* b1 = (const float*)d_in[2];
    const float* Wr = (const float*)d_in[3];
    const float* br = (const float*)d_in[4];
    const float* W2 = (const float*)d_in[5];
    const float* b2 = (const float*)d_in[6];
    float* out = (float*)d_out;

    float* spk1_rec = out;
    float* mem1_rec = out + TBNH;
    float* spk2_rec = out + 2 * TBNH;
    float* mem2_rec = out + 2 * TBNH + TBNO;

    // workspace carve-out (all sizes 256B-aligned)
    char* w = (char*)d_ws;
    _Float16* Xh   = (_Float16*)w; w += (size_t)M_ * NI_ * 2;       //  64 MB
    _Float16* W1h  = (_Float16*)w; w += (size_t)NH_ * NI_ * 2;      //   4 MB
    _Float16* Wrh  = (_Float16*)w; w += (size_t)NH_ * NI_ * 2;      //   4 MB
    _Float16* W2h  = (_Float16*)w; w += (size_t)NO_ * NH_ * 2;      //   1 MB
    float*    cand = (float*)w;    w += (size_t)TBNH * 4;           // 256 MB
    float*    forg = (float*)w;    w += (size_t)TBNH * 4;           // 256 MB
    _Float16* s1h  = (_Float16*)w; w += (size_t)TBNH * 2;           // 128 MB
    float*    inp2 = (float*)w;    w += (size_t)TBNO * 4;           //  32 MB

    // 1) fp32 -> f16 conversions
    {
        long n = M_ * NI_;
        cvt_f32_f16<<<(int)(n / 8 / 256), 256, 0, stream>>>(x, Xh, n);
        n = (long)NH_ * NI_;
        cvt_f32_f16<<<(int)(n / 8 / 256), 256, 0, stream>>>(W1, W1h, n);
        cvt_f32_f16<<<(int)(n / 8 / 256), 256, 0, stream>>>(Wr, Wrh, n);
        n = (long)NO_ * NH_;
        cvt_f32_f16<<<(int)(n / 8 / 256), 256, 0, stream>>>(W2, W2h, n);
    }

    // 2) batched dual GEMM over all T*B rows (WMMA), fused bias+sigmoid/relu
    {
        long waves = (M_ / 64) * (NH_ / 16);   // 65536
        gemm_dual_f16<<<(int)(waves / 8), 256, 0, stream>>>(Xh, W1h, Wrh, b1, br, cand, forg);
    }

    // 3) layer-1 sequential scan, parallel over B*NH elements
    rec1_kernel<<<(B_ * NH_) / 256, 256, 0, stream>>>(cand, forg, spk1_rec, mem1_rec, s1h);

    // 4) batched output GEMM (WMMA), fused bias
    {
        long waves = (M_ / 64) * (NO_ / 16);   // 8192
        gemm_out_f16<<<(int)(waves / 8), 256, 0, stream>>>(s1h, W2h, b2, inp2);
    }

    // 5) layer-2 sequential scan, parallel over B*NO elements
    rec2_kernel<<<(B_ * NO_) / 256, 256, 0, stream>>>(inp2, spk2_rec, mem2_rec);
}
